// NeuralCDE_61710090109758
// MI455X (gfx1250) — compile-verified, matrix operand order checked by reading, not run on hardware
//
#include <hip/hip_runtime.h>
#include <hip/hip_bf16.h>
#include <math.h>

typedef __attribute__((ext_vector_type(16))) _Float16 v16h;
typedef __attribute__((ext_vector_type(8)))  _Float16 v8h;
typedef __attribute__((ext_vector_type(8)))  float    v8f;

#define HDIM 32
#define WID  64
#define CIN  3
#define TLEN 512
#define WAVES_PER_BLOCK 4

// ---- Tsit5 tableau ----
constexpr float T_A21 = 0.161f;
constexpr float T_A31 = -0.008480655492356989f, T_A32 = 0.335480655492357f;
constexpr float T_A41 = 2.8971530571054935f, T_A42 = -6.359448489975075f, T_A43 = 4.3622954328695815f;
constexpr float T_A51 = 5.325864828439257f, T_A52 = -11.748883564062828f, T_A53 = 7.4955393428898365f, T_A54 = -0.09249506636175525f;
constexpr float T_A61 = 5.86145544294642f, T_A62 = -12.92096931784711f, T_A63 = 8.159367898576159f, T_A64 = -0.071584973281401f, T_A65 = -0.028269050394068383f;
constexpr float T_B1 = 0.09646076681806523f, T_B2 = 0.01f, T_B3 = 0.4798896504144996f;
constexpr float T_B4 = 1.379008574103742f, T_B5 = -3.290069515436081f, T_B6 = 2.324710524099774f;

// gfx1250 has a hardware v_tanh_f32 (TRANS pipe, co-executes with WMMA).
// Guard with __has_builtin so compilation never breaks if the name differs.
__device__ __forceinline__ float fast_tanh(float x) {
#if __has_builtin(__builtin_amdgcn_tanhf)
  return __builtin_amdgcn_tanhf(x);
#elif __has_builtin(__builtin_amdgcn_tanh_f32)
  return __builtin_amdgcn_tanh_f32(x);
#else
  return tanhf(x);
#endif
}

// Load an A-fragment (16x32 f16 slice, rows m=lane%16) from a row-major f16 LDS
// buffer. Per CDNA5 ISA A-layout: slots 0..7 = K in [k0+hi*8, k0+hi*8+8),
// slots 8..15 = K in [k0+16+hi*8, ...). Two ds_load_b128 per lane.
__device__ __forceinline__ v16h afrag_load(const _Float16* buf, int rowstride,
                                           int m, int k0, int hi) {
  const v8h* p0 = (const v8h*)(buf + m * rowstride + k0 + hi * 8);
  const v8h* p1 = (const v8h*)(buf + m * rowstride + k0 + 16 + hi * 8);
  v8h lo = *p0;
  v8h hv = *p1;
  v16h r;
#pragma unroll
  for (int i = 0; i < 8; ++i) { r[i] = lo[i]; r[i + 8] = hv[i]; }
  return r;
}

// Load a pre-packed B-fragment (16 halfs per lane, contiguous) from the LDS pool.
__device__ __forceinline__ v16h bfrag_load(const _Float16* pool, int frag, int lane) {
  const v8h* p = (const v8h*)(pool + ((size_t)frag * 32 + lane) * 16);
  v8h a = p[0];
  v8h b = p[1];
  v16h r;
#pragma unroll
  for (int i = 0; i < 8; ++i) { r[i] = a[i]; r[i + 8] = b[i]; }
  return r;
}

// One vector-field evaluation: k = (MLP(ze).reshape(16,H,3) @ dq).
// ze / kout are 16 per-lane f32 regs in WMMA C/D layout:
//   element [t*8+r] = row m=(r+hi*8), hidden h=(16*t + lane%16).
// Biases are folded AFTER the WMMA chain (C operand is inline 0) to avoid
// keeping 14 loop-invariant v8f bias splats resident across the time loop.
__device__ __forceinline__ void vf_eval(
    const float* ze, float* kout, const _Float16* __restrict__ pool,
    _Float16* zbuf, _Float16* h1b, _Float16* h2b,
    const float* bias1, const float* bias2, const float* bias3,
    const float* dqr, int lane) {
  const int n = lane & 15;
  const int hi = lane >> 4;
  const int hi8 = hi * 8;
  const v8f zero = {};

  // D-layout -> row-major f16 [16][32] in LDS (per-wave buffer; DS is in-order per wave)
#pragma unroll
  for (int t = 0; t < 2; ++t)
#pragma unroll
    for (int r = 0; r < 8; ++r)
      zbuf[(r + hi8) * HDIM + t * 16 + n] = (_Float16)ze[t * 8 + r];

  v16h az = afrag_load(zbuf, HDIM, n, 0, hi);

  // Layer 1: h1 = tanh(z @ W1^T + b1)   (4 WMMA, C = inline 0)
#pragma unroll
  for (int t = 0; t < 4; ++t) {
    v16h b = bfrag_load(pool, t, lane);
    v8f acc = __builtin_amdgcn_wmma_f32_16x16x32_f16(false, az, false, b, (short)0, zero, false, false);
#pragma unroll
    for (int r = 0; r < 8; ++r)
      h1b[(r + hi8) * WID + t * 16 + n] = (_Float16)fast_tanh(acc[r] + bias1[t]);
  }

  v16h a0 = afrag_load(h1b, WID, n, 0, hi);
  v16h a1 = afrag_load(h1b, WID, n, 32, hi);

  // Layer 2: h2 = tanh(h1 @ W2^T + b2)  (8 WMMA, K=64 chained through C operand)
#pragma unroll
  for (int t = 0; t < 4; ++t) {
    v16h bk0 = bfrag_load(pool, 4 + t * 2 + 0, lane);
    v8f acc = __builtin_amdgcn_wmma_f32_16x16x32_f16(false, a0, false, bk0, (short)0, zero, false, false);
    v16h bk1 = bfrag_load(pool, 4 + t * 2 + 1, lane);
    acc = __builtin_amdgcn_wmma_f32_16x16x32_f16(false, a1, false, bk1, (short)0, acc, false, false);
#pragma unroll
    for (int r = 0; r < 8; ++r)
      h2b[(r + hi8) * WID + t * 16 + n] = (_Float16)fast_tanh(acc[r] + bias2[t]);
  }

  v16h c0 = afrag_load(h2b, WID, n, 0, hi);
  v16h c1 = afrag_load(h2b, WID, n, 32, hi);

  // Layer 3 split per control channel c: F_c = h2 @ W3_c^T + b3_c  (12 WMMA),
  // then fold dq in D-layout: k[m][h] += (F_c[m][h] + b3_c[h]) * dq[m][c].
#pragma unroll
  for (int i = 0; i < 16; ++i) kout[i] = 0.0f;
#pragma unroll
  for (int c = 0; c < 3; ++c)
#pragma unroll
    for (int th = 0; th < 2; ++th) {
      v16h bk0 = bfrag_load(pool, 12 + (c * 2 + th) * 2 + 0, lane);
      v8f acc = __builtin_amdgcn_wmma_f32_16x16x32_f16(false, c0, false, bk0, (short)0, zero, false, false);
      v16h bk1 = bfrag_load(pool, 12 + (c * 2 + th) * 2 + 1, lane);
      acc = __builtin_amdgcn_wmma_f32_16x16x32_f16(false, c1, false, bk1, (short)0, acc, false, false);
      const float bb = bias3[c * 2 + th];
#pragma unroll
      for (int r = 0; r < 8; ++r)
        kout[th * 8 + r] += (acc[r] + bb) * dqr[c * 8 + r];
    }
}

__global__ __launch_bounds__(32 * WAVES_PER_BLOCK, 1)
void neural_cde_tsit5_wmma(
    const float* __restrict__ times, const float* __restrict__ grad,
    const float* __restrict__ w1, const float* __restrict__ b1,
    const float* __restrict__ w2, const float* __restrict__ b2,
    const float* __restrict__ w3, const float* __restrict__ b3,
    const float* __restrict__ w_enc, const float* __restrict__ b_enc,
    const float* __restrict__ w_ro, const float* __restrict__ b_ro,
    float* __restrict__ out) {
  // Weight B-fragment pool: 24 frags x 32 lanes x 16 halfs = 24 KB (per block)
  __shared__ __align__(16) _Float16 s_wpool[24 * 32 * 16];
  __shared__ __align__(16) _Float16 s_zbuf[WAVES_PER_BLOCK][16 * HDIM];
  __shared__ __align__(16) _Float16 s_h1[WAVES_PER_BLOCK][16 * WID];
  __shared__ __align__(16) _Float16 s_h2[WAVES_PER_BLOCK][16 * WID];
  __shared__ __align__(16) float    s_dq[WAVES_PER_BLOCK][3 * 16];
  __shared__ __align__(16) float    s_red[WAVES_PER_BLOCK][16 * 16];

  const int tid = threadIdx.x;
  const int lane = tid & 31;
  const int wave = tid >> 5;
  const int n = lane & 15;
  const int hi = lane >> 4;
  const int hi8 = hi * 8;

  // ---- pack weight fragments into LDS B-layout (wave 0 only) ----
  if (wave == 0) {
    _Float16 tmp[16];
    // B-layout mirrors A: lane holds column n=lane%16; slot s -> k:
    //   s<8: k = hi*8+s ; s>=8: k = 16+hi*8+(s-8)   (within a 32-wide K chunk)
    // W1 frags 0..3 : B[k][16t+n] = w1[16t+n][k]
#pragma unroll
    for (int t = 0; t < 4; ++t) {
#pragma unroll
      for (int s = 0; s < 16; ++s) {
        int kk = (s < 8) ? (hi8 + s) : (16 + hi8 + (s - 8));
        tmp[s] = (_Float16)w1[(t * 16 + n) * HDIM + kk];
      }
#pragma unroll
      for (int i = 0; i < 16; ++i) s_wpool[((size_t)t * 32 + lane) * 16 + i] = tmp[i];
    }
    // W2 frags 4..11 : B[k][16t+n] = w2[16t+n][32*kf + k]
#pragma unroll
    for (int t = 0; t < 4; ++t)
#pragma unroll
      for (int kf = 0; kf < 2; ++kf) {
#pragma unroll
        for (int s = 0; s < 16; ++s) {
          int kk = (s < 8) ? (hi8 + s) : (16 + hi8 + (s - 8));
          tmp[s] = (_Float16)w2[(t * 16 + n) * WID + kf * 32 + kk];
        }
        int f = 4 + t * 2 + kf;
#pragma unroll
        for (int i = 0; i < 16; ++i) s_wpool[((size_t)f * 32 + lane) * 16 + i] = tmp[i];
      }
    // W3 frags 12..23 : per channel c, row (16*th+n) of W3_c = w3 row ((16*th+n)*3+c)
#pragma unroll
    for (int c = 0; c < 3; ++c)
#pragma unroll
      for (int th = 0; th < 2; ++th)
#pragma unroll
        for (int kf = 0; kf < 2; ++kf) {
          int row = (th * 16 + n) * CIN + c;
#pragma unroll
          for (int s = 0; s < 16; ++s) {
            int kk = (s < 8) ? (hi8 + s) : (16 + hi8 + (s - 8));
            tmp[s] = (_Float16)w3[row * WID + kf * 32 + kk];
          }
          int f = 12 + (c * 2 + th) * 2 + kf;
#pragma unroll
          for (int i = 0; i < 16; ++i) s_wpool[((size_t)f * 32 + lane) * 16 + i] = tmp[i];
        }
  }
  __syncthreads();

  // per-lane scalar bias values (folded post-WMMA)
  float bias1[4], bias2[4], bias3[6];
#pragma unroll
  for (int t = 0; t < 4; ++t) { bias1[t] = b1[t * 16 + n]; bias2[t] = b2[t * 16 + n]; }
#pragma unroll
  for (int c = 0; c < 3; ++c)
#pragma unroll
    for (int th = 0; th < 2; ++th) bias3[c * 2 + th] = b3[(th * 16 + n) * CIN + c];

  const float dt = times[1] - times[0];
  const int b0 = (blockIdx.x * WAVES_PER_BLOCK + wave) * 16;

  // z0 = encoder([1.0]) broadcast over the 16 batch rows (D layout)
  float z[16];
#pragma unroll
  for (int t = 0; t < 2; ++t) {
    float zv = w_enc[t * 16 + n] + b_enc[t * 16 + n];
#pragma unroll
    for (int r = 0; r < 8; ++r) z[t * 8 + r] = zv;
  }

  _Float16* zbuf = s_zbuf[wave];
  _Float16* h1b = s_h1[wave];
  _Float16* h2b = s_h2[wave];
  float* dqlds = s_dq[wave];
  const _Float16* pool = s_wpool;

  float k1[16], k2[16], k3[16], k4[16], k5[16], k6[16], ze[16], dqr[24];

  for (int step = 0; step < TLEN - 1; ++step) {
    // dq[m][c] = dt * gradients[b0+m, step, c]; lane n owns row n (both halves dup)
    {
      const float* gp = grad + ((size_t)(b0 + n) * TLEN + step) * CIN;
      __builtin_prefetch(gp + 8 * CIN, 0, 0);  // stream-ahead hint -> global_prefetch
      float g0 = gp[0] * dt, g1 = gp[1] * dt, g2 = gp[2] * dt;
      dqlds[0 * 16 + n] = g0;
      dqlds[1 * 16 + n] = g1;
      dqlds[2 * 16 + n] = g2;
#pragma unroll
      for (int c = 0; c < 3; ++c) {
        const float4* dp = (const float4*)(dqlds + c * 16 + hi8);  // 32B aligned
        float4 da = dp[0];
        float4 db = dp[1];
        dqr[c * 8 + 0] = da.x; dqr[c * 8 + 1] = da.y;
        dqr[c * 8 + 2] = da.z; dqr[c * 8 + 3] = da.w;
        dqr[c * 8 + 4] = db.x; dqr[c * 8 + 5] = db.y;
        dqr[c * 8 + 6] = db.z; dqr[c * 8 + 7] = db.w;
      }
    }

    vf_eval(z, k1, pool, zbuf, h1b, h2b, bias1, bias2, bias3, dqr, lane);
#pragma unroll
    for (int i = 0; i < 16; ++i) ze[i] = z[i] + T_A21 * k1[i];
    vf_eval(ze, k2, pool, zbuf, h1b, h2b, bias1, bias2, bias3, dqr, lane);
#pragma unroll
    for (int i = 0; i < 16; ++i) ze[i] = z[i] + T_A31 * k1[i] + T_A32 * k2[i];
    vf_eval(ze, k3, pool, zbuf, h1b, h2b, bias1, bias2, bias3, dqr, lane);
#pragma unroll
    for (int i = 0; i < 16; ++i) ze[i] = z[i] + T_A41 * k1[i] + T_A42 * k2[i] + T_A43 * k3[i];
    vf_eval(ze, k4, pool, zbuf, h1b, h2b, bias1, bias2, bias3, dqr, lane);
#pragma unroll
    for (int i = 0; i < 16; ++i)
      ze[i] = z[i] + T_A51 * k1[i] + T_A52 * k2[i] + T_A53 * k3[i] + T_A54 * k4[i];
    vf_eval(ze, k5, pool, zbuf, h1b, h2b, bias1, bias2, bias3, dqr, lane);
#pragma unroll
    for (int i = 0; i < 16; ++i)
      ze[i] = z[i] + T_A61 * k1[i] + T_A62 * k2[i] + T_A63 * k3[i] + T_A64 * k4[i] + T_A65 * k5[i];
    vf_eval(ze, k6, pool, zbuf, h1b, h2b, bias1, bias2, bias3, dqr, lane);
#pragma unroll
    for (int i = 0; i < 16; ++i)
      z[i] += T_B1 * k1[i] + T_B2 * k2[i] + T_B3 * k3[i] + T_B4 * k4[i] + T_B5 * k5[i] + T_B6 * k6[i];
  }

  // readout: sigmoid(z @ w_ro^T + b_ro), reduce over hidden dim via LDS
  {
    float wn = w_ro[n];
    float wn2 = w_ro[16 + n];
    float* red = s_red[wave];
#pragma unroll
    for (int r = 0; r < 8; ++r)
      red[(r + hi8) * 16 + n] = z[r] * wn + z[8 + r] * wn2;
    if (lane < 16) {
      float sum = 0.0f;
#pragma unroll
      for (int j = 0; j < 16; ++j) sum += red[lane * 16 + j];
      float logit = sum + b_ro[0];
      out[b0 + lane] = 1.0f / (1.0f + expf(-logit));
    }
  }
}

extern "C" void kernel_launch(void* const* d_in, const int* in_sizes, int n_in,
                              void* d_out, int out_size, void* d_ws, size_t ws_size,
                              hipStream_t stream) {
  const float* times = (const float*)d_in[0];
  const float* grad  = (const float*)d_in[1];
  const float* w1    = (const float*)d_in[2];
  const float* b1    = (const float*)d_in[3];
  const float* w2    = (const float*)d_in[4];
  const float* b2    = (const float*)d_in[5];
  const float* w3    = (const float*)d_in[6];
  const float* b3    = (const float*)d_in[7];
  const float* w_enc = (const float*)d_in[8];
  const float* b_enc = (const float*)d_in[9];
  const float* w_ro  = (const float*)d_in[10];
  const float* b_ro  = (const float*)d_in[11];
  float* out = (float*)d_out;

  const int Btotal = in_sizes[1] / (TLEN * CIN);   // 16384
  const int tiles = Btotal / 16;                   // 1024 waves
  const int blocks = tiles / WAVES_PER_BLOCK;      // 256 blocks

  neural_cde_tsit5_wmma<<<blocks, 32 * WAVES_PER_BLOCK, 0, stream>>>(
      times, grad, w1, b1, w2, b2, w3, b3, w_enc, b_enc, w_ro, b_ro, out);
}
